// GCN_39702677684864
// MI455X (gfx1250) — compile-verified, matrix operand order checked by reading, not run on hardware
//
#include <hip/hip_runtime.h>

typedef __attribute__((ext_vector_type(16))) _Float16 v16h;
typedef __attribute__((ext_vector_type(8)))  float    v8f;

#define GCN_N 100000
#define GCN_E 640000
#define KPAD  128            // all three layers have in_f <= 128, padded to 128
#define KSTEPS 4             // KPAD / 32

// ---------------------------------------------------------------- utilities
__global__ void gcn_fill_zero(float* __restrict__ p, long long n) {
  long long i = (long long)blockIdx.x * blockDim.x + threadIdx.x;
  if (i < n) p[i] = 0.f;
}

__global__ void gcn_degree(const long long* __restrict__ src,
                           const long long* __restrict__ dst,
                           float* __restrict__ deg_o,
                           float* __restrict__ deg_i, int E) {
  int e = blockIdx.x * blockDim.x + threadIdx.x;
  if (e < E) {
    __hip_atomic_fetch_add(&deg_o[src[e]], 1.f, __ATOMIC_RELAXED, __HIP_MEMORY_SCOPE_AGENT);
    __hip_atomic_fetch_add(&deg_i[dst[e]], 1.f, __ATOMIC_RELAXED, __HIP_MEMORY_SCOPE_AGENT);
  }
}

__global__ void gcn_norm(const float* __restrict__ deg_o,
                         const float* __restrict__ deg_i,
                         float* __restrict__ nrm_o,
                         float* __restrict__ nrm_i, int N) {
  int i = blockIdx.x * blockDim.x + threadIdx.x;
  if (i < N) {
    nrm_o[i] = rsqrtf(fmaxf(deg_o[i], 1.f));
    nrm_i[i] = rsqrtf(fmaxf(deg_i[i], 1.f));
  }
}

// --------------------------------------------------------------- WMMA packing
// WMMA 16-bit A layout (ISA 7.12.2): lane l -> row (l&15), group g = l>>4;
// VGPR v holds K-pair base = (v<4?0:16) + 8*g + 2*(v&3).  Inverting for
// element k within a 32-wide k-step (u = (k&31)>>1):
//   v = ((u>>3)<<2) | (u&3),  g = (u>>2)&1,  half = k&1
// Packed block per (tile, kstep): 32 lanes x 16 halves = 512 halves (1 KB),
// lane's 16 halves contiguous (32 B) -> two global_load_b128 per operand.
__device__ __forceinline__ int wmma_pack_off(int row15, int k) {
  int kk = k & 31;
  int u  = kk >> 1;
  int v  = ((u >> 3) << 2) | (u & 3);
  int g  = (u >> 2) & 1;
  int lane = (g << 4) | row15;
  return (lane << 4) | (v << 1) | (kk & 1);
}

// PA[(mtile*KSTEPS + kstep)*512 + off] = (X[m][k] * nrm[m]) as f16, zero-pad k>=K
__global__ void gcn_pack_a(const float* __restrict__ X, const float* __restrict__ nrm,
                           _Float16* __restrict__ PA, int N, int K) {
  long long idx = (long long)blockIdx.x * blockDim.x + threadIdx.x;
  if (idx >= (long long)N * KPAD) return;
  int m = (int)(idx >> 7);           // idx / 128
  int k = (int)(idx & (KPAD - 1));
  float val = (k < K) ? X[(size_t)m * K + k] * nrm[m] : 0.f;
  int mt = m >> 4, ks = k >> 5;
  size_t dest = ((size_t)(mt * KSTEPS + ks) << 9) + wmma_pack_off(m & 15, k);
  PA[dest] = (_Float16)val;
}

// PB for W [K x OUT]: B column n -> lane (g<<4)|(n&15), same k mapping.
__global__ void gcn_pack_b(const float* __restrict__ W,
                           _Float16* __restrict__ PB, int K, int OUT, int ntn) {
  int OUTP = ntn * 16;
  long long idx = (long long)blockIdx.x * blockDim.x + threadIdx.x;
  if (idx >= (long long)OUTP * KPAD) return;
  int n = (int)(idx >> 7);
  int k = (int)(idx & (KPAD - 1));
  float val = (k < K && n < OUT) ? W[(size_t)k * OUT + n] : 0.f;
  int nt = n >> 4, ks = k >> 5;
  size_t dest = ((size_t)(nt * KSTEPS + ks) << 9) + wmma_pack_off(n & 15, k);
  PB[dest] = (_Float16)val;
}

// ------------------------------------------------- WMMA GEMM: Y = packedA @ packedB
// One wave32 computes one 16x16 tile; K fully unrolled (4 x k32 steps).
__global__ void gcn_gemm_wmma(const _Float16* __restrict__ PA,
                              const _Float16* __restrict__ PB,
                              float* __restrict__ Y,
                              int OUT, int ntn, int total_tiles) {
  int wave = blockIdx.x * (blockDim.x >> 5) + (threadIdx.x >> 5);
  if (wave >= total_tiles) return;       // wave-uniform: EXEC all-1s at WMMA
  int lane = threadIdx.x & 31;
  int mt = wave / ntn;
  int nt = wave - mt * ntn;

  const v16h* pa = (const v16h*)PA + (size_t)mt * (KSTEPS * 32) + lane;
  const v16h* pb = (const v16h*)PB + (size_t)nt * (KSTEPS * 32) + lane;

  v8f acc = {};
#pragma unroll
  for (int ks = 0; ks < KSTEPS; ++ks) {
    v16h a = pa[ks * 32];                // 32B aligned -> 2x global_load_b128
    v16h b = pb[ks * 32];
    acc = __builtin_amdgcn_wmma_f32_16x16x32_f16(
        false, a, false, b, (short)0, acc, false, false);
  }

  int g  = lane >> 4;
  int n  = nt * 16 + (lane & 15);
  if (n < OUT) {
#pragma unroll
    for (int r = 0; r < 8; ++r) {
      int mm = mt * 16 + g * 8 + r;      // C/D layout: VGPR r -> row r + 8*g
      Y[(size_t)mm * OUT + n] = acc[r];
    }
  }
}

// ------------------------------------------ edge scatter-add: AGG[dst] += Y[src]
template <int D, int V>
__global__ void gcn_scatter_t(const long long* __restrict__ src,
                              const long long* __restrict__ dst,
                              const float* __restrict__ Y,
                              float* __restrict__ AGG, int E) {
  constexpr int CH = D / V;              // chunks per edge
  long long idx = (long long)blockIdx.x * blockDim.x + threadIdx.x;
  if (idx >= (long long)E * CH) return;
  int e = (int)(idx / CH);
  int f = (int)(idx - (long long)e * CH) * V;
  int s = (int)src[e];
  int d = (int)dst[e];
  const float* y = Y + (size_t)s * D + f;
  float* a = AGG + (size_t)d * D + f;
#pragma unroll
  for (int i = 0; i < V; ++i)
    __hip_atomic_fetch_add(&a[i], y[i], __ATOMIC_RELAXED, __HIP_MEMORY_SCOPE_AGENT);
}

// ------------------------------------ H = relu(AGG * norm_in[row] + bias[col])
__global__ void gcn_finalize(const float* __restrict__ AGG,
                             const float* __restrict__ nrm_i,
                             const float* __restrict__ bias,
                             float* __restrict__ H, int N, int D) {
  long long idx = (long long)blockIdx.x * blockDim.x + threadIdx.x;
  if (idx >= (long long)N * D) return;
  int r = (int)(idx / D);
  int c = (int)(idx - (long long)r * D);
  float v = AGG[idx] * nrm_i[r] + bias[c];
  H[idx] = v > 0.f ? v : 0.f;
}

// ---------------------------------------------------------------- host side
static inline int cdiv(long long a, long long b) { return (int)((a + b - 1) / b); }

extern "C" void kernel_launch(void* const* d_in, const int* in_sizes, int n_in,
                              void* d_out, int out_size, void* d_ws, size_t ws_size,
                              hipStream_t stream) {
  const float*      x   = (const float*)d_in[0];
  const long long*  ei  = (const long long*)d_in[1];   // int64 edge_index [2 x E]
  const float*      W1  = (const float*)d_in[2];
  const float*      b1  = (const float*)d_in[3];
  const float*      W2  = (const float*)d_in[4];
  const float*      b2  = (const float*)d_in[5];
  const float*      W3  = (const float*)d_in[6];
  const float*      b3  = (const float*)d_in[7];
  float*            out = (float*)d_out;

  const long long* src = ei;
  const long long* dst = ei + GCN_E;
  const int N = GCN_N, E = GCN_E;

  // Workspace (floats): deg_o|deg_i|nrm_o|nrm_i | Y[128N] | AGG[128N] | H[128N] | PB
  float* ws    = (float*)d_ws;
  float* deg_o = ws;
  float* deg_i = deg_o + N;
  float* nrm_o = deg_i + N;
  float* nrm_i = nrm_o + N;
  float* Y     = nrm_i + N;
  float* AGG   = Y   + (size_t)N * 128;
  float* H     = AGG + (size_t)N * 128;
  _Float16* PB = (_Float16*)(H + (size_t)N * 128);     // 128x128 f16 = 32 KB
  _Float16* PA = (_Float16*)AGG;   // PA (25.6 MB) aliases AGG: PA consumed by GEMM
                                   // before AGG is zeroed (stream-ordered)

  // Degrees + norms (recomputed each call for determinism)
  gcn_fill_zero<<<cdiv(2LL * N, 256), 256, 0, stream>>>(deg_o, 2LL * N);
  gcn_degree<<<cdiv(E, 256), 256, 0, stream>>>(src, dst, deg_o, deg_i, E);
  gcn_norm<<<cdiv(N, 256), 256, 0, stream>>>(deg_o, deg_i, nrm_o, nrm_i, N);

  auto layer = [&](const float* in, int K, const float* W, const float* b,
                   int OUT, float* hout) {
    int ntn   = (OUT + 15) / 16;
    int tiles = (N / 16) * ntn;                 // N divisible by 16
    gcn_pack_a<<<cdiv((long long)N * KPAD, 256), 256, 0, stream>>>(in, nrm_o, PA, N, K);
    gcn_pack_b<<<cdiv((long long)ntn * 16 * KPAD, 256), 256, 0, stream>>>(W, PB, K, OUT, ntn);
    gcn_gemm_wmma<<<cdiv(tiles, 4), 128, 0, stream>>>(PA, PB, Y, OUT, ntn, tiles);
    gcn_fill_zero<<<cdiv((long long)N * OUT, 256), 256, 0, stream>>>(AGG, (long long)N * OUT);
    if (OUT == 128)
      gcn_scatter_t<128, 4><<<cdiv((long long)E * 32, 256), 256, 0, stream>>>(src, dst, Y, AGG, E);
    else
      gcn_scatter_t<47, 1><<<cdiv((long long)E * 47, 256), 256, 0, stream>>>(src, dst, Y, AGG, E);
    gcn_finalize<<<cdiv((long long)N * OUT, 256), 256, 0, stream>>>(AGG, nrm_i, b, hout, N, OUT);
  };

  layer(x, 100, W1, b1, 128, H);    // layer 1: 100 -> 128
  layer(H, 128, W2, b2, 128, H);    // layer 2: 128 -> 128
  layer(H, 128, W3, b3,  47, out);  // layer 3: 128 -> 47, writes d_out
}
